// GestureNetwork_77489799954641
// MI455X (gfx1250) — matrix operand order repaired and checked by reading, not compiled, not run on hardware
//
#include <hip/hip_runtime.h>

// ============================================================================
// SLAYER SNN forward for MI455X (gfx1250, wave32, WMMA 16x16x32 f16).
//
//   SP0: pool4 + psp + spike            (8,2,128,128,T)->(B,T,32,32,2) f16
//   SC1: WMMA implicit-GEMM conv 5x5    ->(B,T,32,32,16) f16, then psp+spike
//   SP1: pool2 + psp + spike            ->(B,T,16,16,16)
//   SC2: WMMA implicit-GEMM conv 3x3    ->(B,T,16,16,32), then psp+spike
//   SP2: pool2 + psp + spike            ->(B,T,8,8,32)
//   SF1: WMMA GEMM 512x2048x2400        ->(B*T,512), then psp+spike
//   SF2: VALU GEMM 11x512 + psp+spike -> d_out (B,11,T) f32
//
// f16 intermediates (spikes are {0,1} -> lossless), f32 WMMA accumulation.
// Weights stored m-major [O][KP] so A-fragments are contiguous b128 LDS loads.
// Global->LDS staging uses gfx1250 async-to-LDS ops (ASYNCcnt path) when the
// toolchain exposes them, else plain load/ds_store. ~190 MB workspace.
// ============================================================================

typedef _Float16 h16;
typedef __attribute__((ext_vector_type(16))) _Float16 v16h;
typedef __attribute__((ext_vector_type(8)))  float    v8f;

#define D_EXP  0.36787944117144233f   // exp(-Ts/tau)
#define C_E    2.7182818284590452f    // e*Ts/tau (psp gain)
#define C_R    (-54.365636569180904f) // -2*theta*e*Ts/tauRef
#define THETA_V 10.0f
#define PGAIN   11.0f                 // 1.1*theta pool weight
#define T_STEPS 300
#define BATCH   8

// ---- gfx1250 async global->LDS (guarded; falls back to sync copies) -------
#if defined(__has_builtin)
#if __has_builtin(__builtin_amdgcn_global_load_async_to_lds_b128) && \
    __has_builtin(__builtin_amdgcn_global_load_async_to_lds_b64)
#define HAVE_ASYNC 1
#endif
#endif
#ifndef HAVE_ASYNC
#define HAVE_ASYNC 0
#endif

#if HAVE_ASYNC
// Builtin signature (from compiler diagnostics):
//   b128: (int4 AS1* src, int4 AS3* dst, imm offset, imm cpol)
//   b64 : (int2 AS1* src, int2 AS3* dst, imm offset, imm cpol)
typedef int vi4 __attribute__((ext_vector_type(4)));
typedef int vi2 __attribute__((ext_vector_type(2)));
typedef __attribute__((address_space(1))) vi4 g_vi4;
typedef __attribute__((address_space(3))) vi4 l_vi4;
typedef __attribute__((address_space(1))) vi2 g_vi2;
typedef __attribute__((address_space(3))) vi2 l_vi2;

__device__ __forceinline__ void async_b128(void* l, const void* g) {
  __builtin_amdgcn_global_load_async_to_lds_b128((g_vi4*)g, (l_vi4*)l, 0, 0);
}
__device__ __forceinline__ void async_b64(void* l, const void* g) {
  __builtin_amdgcn_global_load_async_to_lds_b64((g_vi2*)g, (l_vi2*)l, 0, 0);
}
__device__ __forceinline__ void wait_async0() {
#if __has_builtin(__builtin_amdgcn_s_wait_asynccnt)
  __builtin_amdgcn_s_wait_asynccnt(0);
#else
  asm volatile("s_wait_asynccnt 0x0" ::: "memory");
#endif
}
#endif

// One fused step of spike(psp(v)): two 2-state recurrences + threshold.
__device__ __forceinline__ float psp_spike_step(float v, float& p1, float& q1,
                                                float& p2, float& q2) {
  float qn = D_EXP * (q1 + p1);            // psp: q uses OLD p
  p1 = D_EXP * p1 + v;
  q1 = qn;
  float u = C_E * q1;
  float qr = D_EXP * (q2 + p2);            // refractory filter
  q2 = qr;
  float s = (u + C_R * q2 >= THETA_V) ? 1.0f : 0.0f;
  p2 = D_EXP * p2 + s;
  return s;
}

// ---------------------------------------------------------------------------
// Weight repack kernels (f32 -> padded f16, m-major [oc][KP])
// ---------------------------------------------------------------------------
__global__ void prep_w1(const float* __restrict__ W1, h16* __restrict__ w1pad) {
  int i = blockIdx.x * blockDim.x + threadIdx.x;        // 16*64
  if (i >= 16 * 64) return;
  int k = i % 64, oc = i / 64;
  float v = 0.f;
  if (k < 50) {
    int c = k % 2, r = k / 2, kx = r % 5, ky = r / 5;
    v = W1[oc * 50 + c * 25 + ky * 5 + kx];
  }
  w1pad[oc * 64 + k] = (h16)v;
}

__global__ void prep_w2(const float* __restrict__ W2, h16* __restrict__ w2pad) {
  int i = blockIdx.x * blockDim.x + threadIdx.x;        // 32*160
  if (i >= 32 * 160) return;
  int k = i % 160, oc = i / 160;
  float v = 0.f;
  if (k < 144) {
    int c = k % 16, r = k / 16, kx = r % 3, ky = r / 3;
    v = W2[oc * 144 + c * 9 + ky * 3 + kx];
  }
  w2pad[oc * 160 + k] = (h16)v;
}

// Wf1 (512,32,8,8) -> wf1p[o][k], k = (y*8+x)*32 + c  (matches sp2 layout)
__global__ void prep_wf1(const float* __restrict__ Wf1, h16* __restrict__ wf1p) {
  int i = blockIdx.x * blockDim.x + threadIdx.x;        // 512*2048
  if (i >= 512 * 2048) return;
  int o = i >> 11, k = i & 2047;
  int c = k & 31, p = k >> 5, y = p >> 3, x = p & 7;
  wf1p[i] = (h16)Wf1[o * 2048 + c * 64 + y * 8 + x];
}

__global__ void prep_wf2(const float* __restrict__ Wf2, h16* __restrict__ wf2p) {
  int i = blockIdx.x * blockDim.x + threadIdx.x;        // 16*512
  if (i >= 16 * 512) return;
  int o = i >> 9, k = i & 511;
  wf2p[i] = (h16)((o < 11) ? Wf2[o * 512 + k] : 0.f);
}

// ---------------------------------------------------------------------------
// SP0: 4x4 sum-pool (gain 1.1*theta) + psp + spike on raw events.
// s_in (B,2,128,128,T) f32 (T fastest) -> (B,T,32,32,2) f16.
// ---------------------------------------------------------------------------
__global__ void input_pool_scan(const float* __restrict__ s_in,
                                h16* __restrict__ sp0) {
  int idx = blockIdx.x * blockDim.x + threadIdx.x;      // 8*2*32*32
  if (idx >= 16384) return;
  int X = idx & 31, r = idx >> 5;
  int Y = r & 31;  r >>= 5;
  int c = r & 1;   int b = r >> 1;

  const float* base = s_in + (size_t)((b * 2 + c) * 128) * 128 * T_STEPS;
  size_t boff[16];
#pragma unroll
  for (int dy = 0; dy < 4; ++dy)
#pragma unroll
    for (int dx = 0; dx < 4; ++dx)
      boff[dy * 4 + dx] = ((size_t)(4 * Y + dy) * 128 + (4 * X + dx)) * T_STEPS;

  float p1 = 0.f, q1 = 0.f, p2 = 0.f, q2 = 0.f;
  for (int t = 0; t < T_STEPS; ++t) {
    float v = 0.f;
#pragma unroll
    for (int j = 0; j < 16; ++j) v += base[boff[j] + t];
    v *= PGAIN;
    float s = psp_spike_step(v, p1, q1, p2, q2);
    sp0[((size_t)(b * T_STEPS + t) * 1024 + (Y * 32 + X)) * 2 + c] = (h16)s;
  }
}

// ---------------------------------------------------------------------------
// Implicit-GEMM conv via WMMA.  One (b,t) image per workgroup, 8 waves.
// A = weights [O][KP] f16 (LDS, m-major -> b128 fragment loads),
// B = im2col patches from a zero-haloed LDS image.
// ---------------------------------------------------------------------------
template <int C, int O, int KS, int H, int W>
__global__ __launch_bounds__(256) void conv_wmma(const h16* __restrict__ img,
                                                 const h16* __restrict__ wpad,
                                                 h16* __restrict__ out) {
  constexpr int PAD = KS / 2;
  constexpr int HP = H + 2 * PAD, WP = W + 2 * PAD;
  constexpr int K = C * KS * KS;
  constexpr int KP = ((K + 31) / 32) * 32;
  constexpr int KSTEPS = KP / 32;
  constexpr int NT = (H * W) / 16;
  constexpr int MT = O / 16;

  __shared__ __align__(16) h16 limg[HP * WP * C];
  __shared__ __align__(16) h16 lw[O * KP];

  int tid = threadIdx.x;
  size_t bt = blockIdx.x;

  // Weights -> LDS (contiguous; async-to-LDS when available)
#if HAVE_ASYNC
  for (int i = tid; i < (O * KP) / 8; i += 256)
    async_b128(lw + i * 8, wpad + i * 8);
#else
  for (int i = tid; i < (O * KP) / 8; i += 256)
    ((uint4*)lw)[i] = ((const uint4*)wpad)[i];
#endif

  // Zero-haloed image -> LDS
  const h16* src = img + bt * (size_t)(H * W * C);
  for (int i = tid; i < HP * WP * C; i += 256) {
    int c = i % C, r = i / C;
    int px = r % WP, py = r / WP;
    int y = py - PAD, x = px - PAD;
    h16 v = (h16)0.f;
    if (y >= 0 && y < H && x >= 0 && x < W) v = src[(y * W + x) * C + c];
    limg[i] = v;
  }
#if HAVE_ASYNC
  wait_async0();
#endif
  __syncthreads();

  int wv = tid >> 5, lane = tid & 31;
  int L = lane & 15, lh = lane >> 4;
  h16* dst = out + bt * (size_t)(H * W * O);

  for (int tt = wv; tt < NT * MT; tt += 8) {
    int nt = tt % NT, mt = tt / NT;
    int n = nt * 16 + L;
    int y = n / W, x = n % W;
    int m = mt * 16 + L;
    v8f acc = {};
#pragma unroll
    for (int ks = 0; ks < KSTEPS; ++ks) {
      int kbase = ks * 32;
      // A fragment: lane<16 -> K {0..15}, lane>=16 -> {16..31};
      // m-major LDS layout makes both runs contiguous b128 loads.
      union { v16h v; uint4 q[2]; } af;
      const h16* abase = &lw[m * KP + kbase + lh * 16];
      af.q[0] = *(const uint4*)(abase);
      af.q[1] = *(const uint4*)(abase + 8);

      // B fragment (column-major): lane<16 -> K {0..15}, lane>=16 -> {16..31}
      v16h bm;
      if constexpr (C == 16) {
        // 16 consecutive k == one 16-channel pixel -> contiguous b128 loads
        int k0 = kbase + lh * 16;
        int rr = k0 >> 4;
        if (rr > KS * KS - 1) rr = KS * KS - 1;  // padded K: weights are zero
        int kx = rr % KS, ky = rr / KS;
        const h16* pbase = &limg[((y + ky) * WP + (x + kx)) * 16];
        union { v16h v; uint4 q[2]; } bf;
        bf.q[0] = *(const uint4*)(pbase);
        bf.q[1] = *(const uint4*)(pbase + 8);
        bm = bf.v;
      } else {
#pragma unroll
        for (int e = 0; e < 16; ++e) {
          int kb = kbase + e + lh * 16;
          int kc = (kb < K) ? kb : 0;  // padded K: weights are zero
          int cc = kc % C, rr = kc / C;
          int kx = rr % KS, ky = rr / KS;
          bm[e] = limg[((y + ky) * WP + (x + kx)) * C + cc];
        }
      }
      acc = __builtin_amdgcn_wmma_f32_16x16x32_f16(false, af.v, false, bm,
                                                   (short)0, acc, false, false);
    }
    // D: VGPR j -> (M = mt*16 + 8*lh + j, N = n); contiguous oc -> b128 store
    union { h16 h[8]; uint4 u; } pk;
#pragma unroll
    for (int j = 0; j < 8; ++j) pk.h[j] = (h16)acc[j];
    *(uint4*)(&dst[(size_t)n * O + mt * 16 + 8 * lh]) = pk.u;
  }
}

// ---------------------------------------------------------------------------
// psp + spike scan, layout (B,T,HWC) f16. One thread per neuron.
// ---------------------------------------------------------------------------
__global__ void psp_spike_scan(const h16* __restrict__ u, h16* __restrict__ s,
                               int HWC) {
  int idx = blockIdx.x * blockDim.x + threadIdx.x;
  if (idx >= BATCH * HWC) return;
  int b = idx / HWC, r = idx % HWC;
  size_t base = (size_t)b * T_STEPS * HWC + r;
  float p1 = 0.f, q1 = 0.f, p2 = 0.f, q2 = 0.f;
  for (int t = 0; t < T_STEPS; ++t) {
    size_t off = base + (size_t)t * HWC;
    float v = (float)u[off];
    s[off] = (h16)psp_spike_step(v, p1, q1, p2, q2);
  }
}

// 2x2 sum-pool (gain 1.1*theta) + psp + spike. (B,T,H,W,C) -> (B,T,H/2,W/2,C)
__global__ void pool2_psp_spike(const h16* __restrict__ sin,
                                h16* __restrict__ sout, int H, int W, int C) {
  int H2 = H >> 1, W2 = W >> 1;
  int total = BATCH * H2 * W2 * C;
  int idx = blockIdx.x * blockDim.x + threadIdx.x;
  if (idx >= total) return;
  int c = idx % C, r = idx / C;
  int X = r % W2;  r /= W2;
  int Y = r % H2;  int b = r / H2;
  float p1 = 0.f, q1 = 0.f, p2 = 0.f, q2 = 0.f;
  for (int t = 0; t < T_STEPS; ++t) {
    size_t bt = (size_t)(b * T_STEPS + t);
    const h16* base = sin + bt * (size_t)(H * W * C);
    float v = (float)base[((2 * Y) * W + 2 * X) * C + c] +
              (float)base[((2 * Y) * W + 2 * X + 1) * C + c] +
              (float)base[((2 * Y + 1) * W + 2 * X) * C + c] +
              (float)base[((2 * Y + 1) * W + 2 * X + 1) * C + c];
    v *= PGAIN;
    float s = psp_spike_step(v, p1, q1, p2, q2);
    sout[bt * (size_t)(H2 * W2 * C) + (Y * W2 + X) * C + c] = (h16)s;
  }
}

// ---------------------------------------------------------------------------
// SF1 GEMM: out[n][o] = sum_k B[n][k]*A[o][k];  M=512, K=2048, N=2400.
// Block = 128(M) x 32(N), 8 waves, 64 k-steps; async tile staging + prefetch.
// ---------------------------------------------------------------------------
__global__ __launch_bounds__(256) void fc1_wmma(const h16* __restrict__ Bmat,
                                                const h16* __restrict__ Amat,
                                                h16* __restrict__ out) {
  __shared__ __align__(16) h16 As[128 * 32];
  __shared__ __align__(16) h16 Bs[32 * 32];
  int tid = threadIdx.x;
  int nb = blockIdx.x * 32;
  int mb = blockIdx.y * 128;
  int wv = tid >> 5, lane = tid & 31;
  int L = lane & 15, lh = lane >> 4;
  v8f acc0 = {}, acc1 = {};
  int arow = tid >> 1, apart = (tid & 1) * 16;
  int brow = tid >> 3, bpart = (tid & 7) * 4;

  const h16* asrc0 = Amat + (size_t)(mb + arow) * 2048 + apart;
  const h16* bsrc0 = Bmat + (size_t)(nb + brow) * 2048 + bpart;
  h16* adst = As + arow * 32 + apart;
  h16* bdst = Bs + brow * 32 + bpart;

  for (int kb = 0; kb < 64; ++kb) {
    int kbase = kb * 32;
#if HAVE_ASYNC
    async_b128(adst, asrc0 + kbase);
    async_b128(adst + 8, asrc0 + kbase + 8);
    async_b64(bdst, bsrc0 + kbase);
    wait_async0();
#else
    ((uint4*)adst)[0] = ((const uint4*)(asrc0 + kbase))[0];
    ((uint4*)adst)[1] = ((const uint4*)(asrc0 + kbase))[1];
    *(uint2*)bdst = *(const uint2*)(bsrc0 + kbase);
#endif
    __syncthreads();
    if (kb < 63) {  // prefetch next K tile into L2/L0
      __builtin_prefetch(asrc0 + kbase + 32, 0, 1);
      __builtin_prefetch(bsrc0 + kbase + 32, 0, 1);
    }

    v16h a, b0, b1;
#pragma unroll
    for (int e = 0; e < 16; ++e) {
      int ka = e + lh * 16;
      a[e] = As[(wv * 16 + L) * 32 + ka];
      int kk = e + lh * 16;
      b0[e] = Bs[L * 32 + kk];
      b1[e] = Bs[(16 + L) * 32 + kk];
    }
    acc0 = __builtin_amdgcn_wmma_f32_16x16x32_f16(false, a, false, b0,
                                                  (short)0, acc0, false, false);
    acc1 = __builtin_amdgcn_wmma_f32_16x16x32_f16(false, a, false, b1,
                                                  (short)0, acc1, false, false);
    __syncthreads();
  }
  union { h16 h[8]; uint4 u; } p0, p1;
#pragma unroll
  for (int j = 0; j < 8; ++j) {
    p0.h[j] = (h16)acc0[j];
    p1.h[j] = (h16)acc1[j];
  }
  int o = mb + wv * 16 + 8 * lh;
  *(uint4*)(out + (size_t)(nb + L) * 512 + o) = p0.u;
  *(uint4*)(out + (size_t)(nb + 16 + L) * 512 + o) = p1.u;
}

// SF2: M=11 underfills a tensor tile -> plain VALU dot, 38400 threads.
__global__ void fc2_kernel(const h16* __restrict__ sf1,
                           const h16* __restrict__ wf2p,
                           h16* __restrict__ uf2) {
  int idx = blockIdx.x * blockDim.x + threadIdx.x;      // n*16 + o
  if (idx >= 2400 * 16) return;
  int o = idx & 15, n = idx >> 4;
  const h16* x = sf1 + (size_t)n * 512;
  const h16* w = wf2p + o * 512;
  float s = 0.f;
  for (int k = 0; k < 512; ++k) s += (float)x[k] * (float)w[k];
  uf2[idx] = (h16)s;
}

// Final psp+spike; writes (B,11,T) f32 to d_out.
__global__ void final_scan(const h16* __restrict__ uf2, float* __restrict__ out) {
  int idx = blockIdx.x * blockDim.x + threadIdx.x;
  if (idx >= BATCH * 11) return;
  int o = idx % 11, b = idx / 11;
  float p1 = 0.f, q1 = 0.f, p2 = 0.f, q2 = 0.f;
  for (int t = 0; t < T_STEPS; ++t) {
    float v = (float)uf2[(size_t)(b * T_STEPS + t) * 16 + o];
    out[(size_t)(b * 11 + o) * T_STEPS + t] = psp_spike_step(v, p1, q1, p2, q2);
  }
}

// ---------------------------------------------------------------------------
extern "C" void kernel_launch(void* const* d_in, const int* in_sizes, int n_in,
                              void* d_out, int out_size, void* d_ws,
                              size_t ws_size, hipStream_t stream) {
  const float* s_in = (const float*)d_in[0];
  const float* W1   = (const float*)d_in[1];
  const float* W2   = (const float*)d_in[2];
  const float* Wf1  = (const float*)d_in[3];
  const float* Wf2  = (const float*)d_in[4];
  float* out = (float*)d_out;
  h16* ws = (h16*)d_ws;

  // Workspace plan (offsets in f16 elements, total ~190 MB, 16B aligned):
  const size_t SP0 = 0;                          // (B,T,32,32,2)   4,915,200
  const size_t RB  = SP0 + 4915200ull;           // region B        39,321,600
  const size_t RC  = RB + 39321600ull;           // region C        39,321,600
  const size_t SP1 = RC + 39321600ull;           // (B,T,16,16,16)  9,830,400
  const size_t SP2 = SP1 + 9830400ull;           // (B,T,8,8,32)    4,915,200
  const size_t W1P = SP2 + 4915200ull;           // 16*64
  const size_t W2P = W1P + 1024ull;              // 32*160
  const size_t WF1P = W2P + 5120ull;             // 512*2048
  const size_t WF2P = WF1P + 1048576ull;         // 16*512
  // Aliases within regions (lifetimes do not overlap):
  const size_t U1  = RB;                         // conv1 out (B,T,32,32,16)
  const size_t S1  = RC;                         // conv1 spikes
  const size_t U2  = RB;                         // conv2 out (B,T,16,16,32)
  const size_t S2  = RC;                         // conv2 spikes
  const size_t UF1 = RB + 19660800ull;           // (2400,512)
  const size_t SF1 = RC + 19660800ull;           // (2400,512)
  const size_t UF2 = RB + 19660800ull + 1228800ull; // (2400,16)

  // Weight repack
  prep_w1<<<4, 256, 0, stream>>>(W1, ws + W1P);
  prep_w2<<<20, 256, 0, stream>>>(W2, ws + W2P);
  prep_wf1<<<4096, 256, 0, stream>>>(Wf1, ws + WF1P);
  prep_wf2<<<32, 256, 0, stream>>>(Wf2, ws + WF2P);

  // SP0
  input_pool_scan<<<64, 256, 0, stream>>>(s_in, ws + SP0);
  // SC1
  conv_wmma<2, 16, 5, 32, 32><<<2400, 256, 0, stream>>>(ws + SP0, ws + W1P, ws + U1);
  psp_spike_scan<<<512, 256, 0, stream>>>(ws + U1, ws + S1, 16384);
  // SP1
  pool2_psp_spike<<<128, 256, 0, stream>>>(ws + S1, ws + SP1, 32, 32, 16);
  // SC2
  conv_wmma<16, 32, 3, 16, 16><<<2400, 256, 0, stream>>>(ws + SP1, ws + W2P, ws + U2);
  psp_spike_scan<<<256, 256, 0, stream>>>(ws + U2, ws + S2, 8192);
  // SP2
  pool2_psp_spike<<<64, 256, 0, stream>>>(ws + S2, ws + SP2, 16, 16, 32);
  // SF1
  fc1_wmma<<<dim3(75, 4), 256, 0, stream>>>(ws + SP2, ws + WF1P, ws + UF1);
  psp_spike_scan<<<16, 256, 0, stream>>>(ws + UF1, ws + SF1, 512);
  // SF2
  fc2_kernel<<<150, 256, 0, stream>>>(ws + SF1, ws + WF2P, ws + UF2);
  final_scan<<<1, 128, 0, stream>>>(ws + UF2, out);
}
